// OccupancyFlowNetFineTuningHarness_74131135529676
// MI455X (gfx1250) — matrix operand order repaired and checked by reading, not compiled, not run on hardware
//
#include <hip/hip_runtime.h>
#include <hip/hip_bf16.h>

typedef __attribute__((ext_vector_type(16))) _Float16 v16h;
typedef __attribute__((ext_vector_type(8)))  float    v8f;

#define S_     16
#define A_     256
#define T_     91
#define H_     256
#define C_     128
#define N_     (A_ * T_)     // 23296
#define DT_    0.1f
#define THIST_ 11
#define DTRAJ_ 8
#define PROAD_ 4096
#define DROAD_ 16

// workspace layout (floats)
#define WS_CTX   0            // S*C = 2048
#define WS_BIAS  2048         // S*H = 4096
#define WS_SE    6144         // S
#define WS_CNT   6160         // S
#define WS_OCC   6176         // 1

// ---------------------------------------------------------------- init
__global__ void k_init(float* ws) {
    int i = threadIdx.x;
    if (i < 33) ws[WS_SE + i] = 0.0f;
}

// ---------------------------------------------------- ctx per scene
// grid = S blocks of 256 threads; block s computes ctx[s][0..127]
__global__ void k_ctx(const float* __restrict__ traj,
                      const float* __restrict__ road,
                      const unsigned char* __restrict__ agent_mask,
                      const float* __restrict__ Wctx,
                      const float* __restrict__ Wr,
                      float* __restrict__ ws) {
    __shared__ float s_tp[DTRAJ_];
    __shared__ float s_rm[DROAD_];
    __shared__ float s_cnt;
    int s = blockIdx.x;
    int a = threadIdx.x;
    if (a < DTRAJ_) s_tp[a] = 0.0f;
    if (a < DROAD_) s_rm[a] = 0.0f;
    if (a == 0) s_cnt = 0.0f;
    __syncthreads();

    // trajectory pool: sum over (a, t_hist) per dim, masked
    float loc[DTRAJ_];
    for (int d = 0; d < DTRAJ_; ++d) loc[d] = 0.0f;
    float m = agent_mask[s * A_ + a] ? 1.0f : 0.0f;
    if (m != 0.0f) {
        const float* tp = traj + ((size_t)(s * A_ + a)) * THIST_ * DTRAJ_;
        for (int th = 0; th < THIST_; ++th)
            for (int d = 0; d < DTRAJ_; ++d)
                loc[d] += tp[th * DTRAJ_ + d];
    }
    for (int d = 0; d < DTRAJ_; ++d) atomicAdd(&s_tp[d], loc[d]);
    atomicAdd(&s_cnt, m);

    // road map mean over P_ROAD points
    float locr[DROAD_];
    for (int r = 0; r < DROAD_; ++r) locr[r] = 0.0f;
    for (int p = a; p < PROAD_; p += blockDim.x) {
        const float* rp = road + ((size_t)(s * PROAD_ + p)) * DROAD_;
        for (int r = 0; r < DROAD_; ++r) locr[r] += rp[r];
    }
    for (int r = 0; r < DROAD_; ++r) atomicAdd(&s_rm[r], locr[r]);
    __syncthreads();

    int c = threadIdx.x;
    if (c < C_) {
        float denom = s_cnt * (float)THIST_;
        float acc = 0.0f;
        for (int d = 0; d < DTRAJ_; ++d)
            acc += (s_tp[d] / denom) * Wctx[d * C_ + c];
        for (int r = 0; r < DROAD_; ++r)
            acc += (s_rm[r] * (1.0f / PROAD_)) * Wr[r * C_ + c];
        ws[WS_CTX + s * C_ + c] = tanhf(acc);
    }
}

// ----------------------------------- bias = ctx @ Wc2 via WMMA (f16 in, f32 acc)
// one block, 16 waves; wave w owns N-tile [16w, 16w+16); M = 16 scenes; K = 128
__global__ void k_bias_wmma(const float* __restrict__ Wc2,
                            float* __restrict__ ws) {
    const float* ctx = ws + WS_CTX;
    float* bias = ws + WS_BIAS;
    int lane = threadIdx.x & 31;
    int wave = threadIdx.x >> 5;
    int n0 = wave * 16;
    int lh = lane >> 4;        // lane-half (0 or 1)
    int mm = lane & 15;        // A-row (scene) / B-col within tile

    v8f acc = {};
    #pragma unroll
    for (int k0 = 0; k0 < C_; k0 += 32) {
        v16h a, b;
        #pragma unroll
        for (int e = 0; e < 16; ++e) {
            // A 16x32 f16 layout (ISA 7.12.2): k = e + 8*lh + (e>=8 ? 8 : 0)
            int ka = k0 + e + 8 * lh + ((e >= 8) ? 8 : 0);
            a[e] = (_Float16)ctx[mm * C_ + ka];
            // B 32x16 f16 layout: same k order as A: k = e + 8*lh + (e>=8 ? 8 : 0)
            int kb = k0 + e + 8 * lh + ((e >= 8) ? 8 : 0);
            b[e] = (_Float16)Wc2[kb * H_ + n0 + mm];
        }
        acc = __builtin_amdgcn_wmma_f32_16x16x32_f16(
            /*neg_a=*/false, a, /*neg_b=*/false, b,
            /*c_mod=*/(short)0, acc, /*reuse_a=*/false, /*reuse_b=*/false);
    }
    #pragma unroll
    for (int r = 0; r < 8; ++r) {
        int scene = r + 8 * lh;                 // D: VGPR r -> M = r (+8 for hi lanes)
        bias[scene * H_ + n0 + mm] = acc[r];
    }
}

// ------------------------------------------------- per-thread flow eval (full H)
__device__ __forceinline__ void flow_eval(float px, float py, float t,
                                          const float* sWp0, const float* sWp1,
                                          const float* swt, const float* sWo0,
                                          const float* sWo1, const float* sb,
                                          float& f0, float& f1) {
    float a0 = 0.0f, a1 = 0.0f;
    for (int j = 0; j < H_; ++j) {
        float h = tanhf(px * sWp0[j] + py * sWp1[j] + t * swt[j] + sb[j]);
        a0 += h * sWo0[j];
        a1 += h * sWo1[j];
    }
    f0 = a0; f1 = a1;
}

// ------------------------------------------------- flow loss over all points
// grid = S*T blocks of 256 threads; block handles scene s, time-chunk tt
__global__ void k_flow_loss(const float* __restrict__ times,
                            const float* __restrict__ positions,
                            const float* __restrict__ velocities,
                            const unsigned char* __restrict__ ff_mask,
                            const float* __restrict__ Wp,
                            const float* __restrict__ wt,
                            const float* __restrict__ Wo,
                            float* __restrict__ ws) {
    __shared__ float sWp0[H_], sWp1[H_], swt[H_], sWo0[H_], sWo1[H_], sb[H_];
    int s  = blockIdx.x / T_;
    int tt = blockIdx.x % T_;
    int j  = threadIdx.x;
    sWp0[j] = Wp[0 * H_ + j];
    sWp1[j] = Wp[1 * H_ + j];
    swt[j]  = wt[j];
    sWo0[j] = Wo[j * 2 + 0];
    sWo1[j] = Wo[j * 2 + 1];
    sb[j]   = ws[WS_BIAS + s * H_ + j];
    __syncthreads();

    int n = tt * A_ + threadIdx.x;          // n = t*A + a
    size_t idx = (size_t)s * N_ + n;
    float px = positions[idx * 2 + 0];
    float py = positions[idx * 2 + 1];
    float t  = times[idx];
    float vx = velocities[idx * 2 + 0];
    float vy = velocities[idx * 2 + 1];
    float msk = ff_mask[idx] ? 1.0f : 0.0f;

    float f0, f1;
    flow_eval(px, py, t, sWp0, sWp1, swt, sWo0, sWo1, sb, f0, f1);
    float se = ((f0 - vx) * (f0 - vx) + (f1 - vy) * (f1 - vy)) * msk;

    atomicAdd(&ws[WS_SE + s], se);
    atomicAdd(&ws[WS_CNT + s], msk);
}

// ------------------------------------------------- rollout (occupancy loss)
// ONE WAVE32 PER AGENT: lane owns 8 hidden units (stride 32), butterfly-reduce.
// grid = S*A/8 blocks of 256 threads (8 waves = 8 agents per block).
__global__ void k_rollout(const float* __restrict__ positions,
                          const float* __restrict__ Wp,
                          const float* __restrict__ wt,
                          const float* __restrict__ Wo,
                          float* __restrict__ ws) {
    __shared__ float sWp0[H_], sWp1[H_], swt[H_], sWo0[H_], sWo1[H_], sb[H_];
    const int blocksPerScene = A_ / 8;               // 32
    int s = blockIdx.x / blocksPerScene;
    int j = threadIdx.x;
    sWp0[j] = Wp[0 * H_ + j];
    sWp1[j] = Wp[1 * H_ + j];
    swt[j]  = wt[j];
    sWo0[j] = Wo[j * 2 + 0];
    sWo1[j] = Wo[j * 2 + 1];
    sb[j]   = ws[WS_BIAS + s * H_ + j];
    __syncthreads();

    int lane = threadIdx.x & 31;
    int wave = threadIdx.x >> 5;
    int a = (blockIdx.x % blocksPerScene) * 8 + wave;   // agent id

    // stage this lane's 8 hidden-unit weights into registers (LDS -> VGPR once)
    float wP0[8], wP1[8], wT[8], wO0[8], wO1[8], wB[8];
    #pragma unroll
    for (int i = 0; i < 8; ++i) {
        int h = lane + 32 * i;          // bank-conflict-free stride
        wP0[i] = sWp0[h]; wP1[i] = sWp1[h]; wT[i] = swt[h];
        wO0[i] = sWo0[h]; wO1[i] = sWo1[h]; wB[i] = sb[h];
    }

    // pos_r[s][t][a][k] = positions[((s*T + t)*A + a)*2 + k]
    size_t base0 = ((size_t)(s * T_ + 0) * A_ + a) * 2;
    float sx = positions[base0 + 0];
    float sy = positions[base0 + 1];

    float acc = 0.0f;
    for (int t = 0; t < T_; ++t) {
        size_t bi = ((size_t)(s * T_ + t) * A_ + a) * 2;
        // prefetch next timestep's reference position (global_prefetch_b8)
        __builtin_prefetch(&positions[bi + (size_t)A_ * 2], 0, 3);
        float rx = positions[bi + 0];
        float ry = positions[bi + 1];
        if (lane == 0)
            acc += fabsf(rx - sx) + fabsf(ry - sy);

        float tv = (float)t * DT_;
        float a0 = 0.0f, a1 = 0.0f;
        #pragma unroll
        for (int i = 0; i < 8; ++i) {
            float h = tanhf(sx * wP0[i] + sy * wP1[i] + tv * wT[i] + wB[i]);
            a0 += h * wO0[i];
            a1 += h * wO1[i];
        }
        // wave32 butterfly reduction: all lanes end with the full dot products
        #pragma unroll
        for (int m = 16; m >= 1; m >>= 1) {
            a0 += __shfl_xor(a0, m, 32);
            a1 += __shfl_xor(a1, m, 32);
        }
        sx += DT_ * a0;
        sy += DT_ * a1;
    }
    if (lane == 0)
        atomicAdd(&ws[WS_OCC], acc);
}

// ------------------------------------------------- finalize two scalars
__global__ void k_finalize(const float* __restrict__ ws, float* __restrict__ out) {
    if (threadIdx.x == 0) {
        float fl = 0.0f;
        for (int s = 0; s < S_; ++s)
            fl += ws[WS_SE + s] / ws[WS_CNT + s];
        out[0] = fl / (float)S_;
        out[1] = ws[WS_OCC] / ((float)S_ * T_ * A_ * 2.0f);
    }
}

extern "C" void kernel_launch(void* const* d_in, const int* in_sizes, int n_in,
                              void* d_out, int out_size, void* d_ws, size_t ws_size,
                              hipStream_t stream) {
    const float* times      = (const float*)d_in[0];
    const float* positions  = (const float*)d_in[1];
    const float* velocities = (const float*)d_in[2];
    const float* road_map   = (const float*)d_in[3];
    const float* traj       = (const float*)d_in[4];
    // d_in[5] agent_ids: unused by the reference math
    const unsigned char* agent_mask = (const unsigned char*)d_in[6];
    const unsigned char* ff_mask    = (const unsigned char*)d_in[7];
    const float* Wp   = (const float*)d_in[8];
    const float* wt   = (const float*)d_in[9];
    const float* Wc2  = (const float*)d_in[10];
    const float* Wo   = (const float*)d_in[11];
    const float* Wctx = (const float*)d_in[12];
    const float* Wr   = (const float*)d_in[13];
    float* ws  = (float*)d_ws;
    float* out = (float*)d_out;

    k_init<<<1, 64, 0, stream>>>(ws);
    k_ctx<<<S_, 256, 0, stream>>>(traj, road_map, agent_mask, Wctx, Wr, ws);
    k_bias_wmma<<<1, 512, 0, stream>>>(Wc2, ws);
    k_flow_loss<<<S_ * T_, 256, 0, stream>>>(times, positions, velocities,
                                             ff_mask, Wp, wt, Wo, ws);
    k_rollout<<<(S_ * A_) / 8, 256, 0, stream>>>(positions, Wp, wt, Wo, ws);
    k_finalize<<<1, 32, 0, stream>>>(ws, out);
}
